// RoPEMultiHeadAttention_21174188769461
// MI455X (gfx1250) — matrix operand order repaired and checked by reading, not compiled, not run on hardware
//
#include <hip/hip_runtime.h>
#include <hip/hip_bf16.h>
#include <math.h>

// ---------------------------------------------------------------------------
// RoPE + causal attention, full embed dim (no head split), MI455X (gfx1250).
// bf16 WMMA (16x16x32), flash-attention online softmax, async global->LDS
// tile staging (ASYNCcnt), software-pipelined fragment loads.
// ---------------------------------------------------------------------------

typedef __attribute__((ext_vector_type(16))) __bf16 v16bf;
typedef __attribute__((ext_vector_type(8)))  float  v8f;

#define SEQ_LEN 4096
#define EMBED   512
#define BATCH   4
#define NROWS   (BATCH * SEQ_LEN)      // 16384
#define SCALE   0.125f                 // 1/sqrt(HEAD_DIM=64) per reference

#define BM 64                          // query rows per block
#define BN 64                          // key rows per tile
#define LDQ (EMBED + 8)                // LDS stride for 512-wide bf16 tiles
#define LDP (BN + 8)                   // LDS stride for P tile (72)
#define LDV (BN + 8)                   // LDS stride for V^T tile (72)

union frag16 { v16bf v; uint4 q[2]; };
union bfpair { __bf16 h[2]; unsigned u; };

// ---------------- CDNA5 async global->LDS copy (ASYNCcnt) ------------------
// Per-lane: LDS[vdst] = MEM[vaddr], 16 bytes. Tracked with ASYNCcnt.
__device__ __forceinline__ void async_copy_b128(void* lds_dst, const void* gsrc) {
  unsigned laddr = (unsigned)(size_t)lds_dst;   // flat LDS addr low 32b = offset
  asm volatile("global_load_async_to_lds_b128 %0, %1, off"
               :: "v"(laddr), "v"(gsrc) : "memory");
}
__device__ __forceinline__ void async_wait0() {
  asm volatile("s_wait_asynccnt 0x0" ::: "memory");
}

// ---------------- WMMA fragment helpers (wave32 layouts per CDNA5 ISA) -----

// A-matrix 16x32 bf16 from row-major [m][k] LDS tile.
__device__ __forceinline__ v16bf lds_load_a(const __bf16* base, int ld,
                                            int row0, int k0, int lane) {
  const int m = lane & 15, half = lane >> 4;
  const __bf16* p = base + (size_t)(row0 + m) * ld + k0 + 8 * half;
  frag16 f;
  f.q[0] = *(const uint4*)p;          // K = 8*half + 0..7
  f.q[1] = *(const uint4*)(p + 16);   // K = 16 + 8*half + 0..7
  return f.v;
}

// B-matrix 32x16 bf16 sourced from a row-major [n][k] LDS tile (transposed
// operand; e.g. K^T from the K tile, or V from the V^T tile).
__device__ __forceinline__ v16bf lds_load_bt(const __bf16* base, int ld,
                                             int n0, int k0, int lane) {
  const int n = lane & 15, half = lane >> 4;
  const __bf16* p = base + (size_t)(n0 + n) * ld + k0 + 8 * half;
  frag16 f;
  f.q[0] = *(const uint4*)p;          // K = 8*half + 0..7
  f.q[1] = *(const uint4*)(p + 16);   // K = 16 + 8*half + 0..7
  return f.v;
}

__device__ __forceinline__ v8f wmma_bf16(v16bf a, v16bf b, v8f c) {
  return __builtin_amdgcn_wmma_f32_16x16x32_bf16(false, a, false, b,
                                                 (short)0, c, false, false);
}

__device__ __forceinline__ v8f zero_v8f() {
  v8f z = {0.f, 0.f, 0.f, 0.f, 0.f, 0.f, 0.f, 0.f};
  return z;
}

// ---------------- Phase 1: QKV projection (x @ W + b) ----------------------
// grid (NROWS/64, EMBED/64, 3); block 128 (4 waves); 64x64 output tile.
// sel 0/1 -> fp32 raw q/k (RoPE applied later in fp32);
// sel 2   -> bf16 V stored TRANSPOSED in global: vT[b][d][s].
__global__ __launch_bounds__(128) void rope_attn_qkv_gemm(
    const float* __restrict__ x,
    const float* __restrict__ Wq, const float* __restrict__ bq,
    const float* __restrict__ Wk, const float* __restrict__ bk,
    const float* __restrict__ Wv, const float* __restrict__ bv,
    float* __restrict__ qraw, float* __restrict__ kraw,
    __bf16* __restrict__ vT) {
  __shared__ __bf16 sX[64][40];    // 64 rows x 32 k
  __shared__ __bf16 sWt[64][40];   // transposed: 64 n x 32 k

  const int row0 = blockIdx.x * 64;
  const int c0   = blockIdx.y * 64;
  const int sel  = blockIdx.z;
  const float* W    = (sel == 0) ? Wq : ((sel == 1) ? Wk : Wv);
  const float* bias = (sel == 0) ? bq : ((sel == 1) ? bk : bv);

  const int tid = threadIdx.x, lane = tid & 31, w = tid >> 5;
  v8f acc[4];
#pragma unroll
  for (int c = 0; c < 4; ++c) acc[c] = zero_v8f();

  for (int k0 = 0; k0 < EMBED; k0 += 32) {
    __syncthreads();
    // x tile: float4 loads, packed bf16 pairs, 8-byte LDS stores.
    for (int i = tid; i < 64 * 8; i += 128) {
      int r = i >> 3, c4 = (i & 7) * 4;
      float4 xv = *(const float4*)&x[(size_t)(row0 + r) * EMBED + k0 + c4];
      bfpair p0, p1;
      p0.h[0] = (__bf16)xv.x; p0.h[1] = (__bf16)xv.y;
      p1.h[0] = (__bf16)xv.z; p1.h[1] = (__bf16)xv.w;
      uint2 st; st.x = p0.u; st.y = p1.u;
      *(uint2*)&sX[r][c4] = st;
    }
    // W tile transposed: coalesced global read, scattered b16 LDS write.
    for (int i = tid; i < 32 * 64; i += 128) {
      int r = i >> 6, c = i & 63;   // r = k, c = n
      sWt[c][r] = (__bf16)W[(size_t)(k0 + r) * EMBED + c0 + c];
    }
    __syncthreads();

    v16bf a = lds_load_a(&sX[0][0], 40, 16 * w, 0, lane);
#pragma unroll
    for (int c = 0; c < 4; ++c) {
      v16bf b = lds_load_bt(&sWt[0][0], 40, 16 * c, 0, lane);
      acc[c] = wmma_bf16(a, b, acc[c]);
    }
  }

  const int n = lane & 15, half = lane >> 4;
#pragma unroll
  for (int c = 0; c < 4; ++c) {
    const int col = c0 + 16 * c + n;
    const float bb = bias[col];
#pragma unroll
    for (int v = 0; v < 8; ++v) {
      const int row = row0 + 16 * w + v + 8 * half;   // row = b*S + s
      const float val = acc[c][v] + bb;
      if (sel == 0)      qraw[(size_t)row * EMBED + col] = val;
      else if (sel == 1) kraw[(size_t)row * EMBED + col] = val;
      else {
        const int bb_ = row >> 12;                    // row / SEQ_LEN
        const int ss  = row & (SEQ_LEN - 1);
        vT[(size_t)bb_ * EMBED * SEQ_LEN + (size_t)col * SEQ_LEN + ss] =
            (__bf16)val;
      }
    }
  }
}

// ---------------- Phase 1b: RoPE (fp32) -> bf16 Qr, Kr ---------------------
__global__ __launch_bounds__(256) void rope_attn_rope(
    const float* __restrict__ qraw, const float* __restrict__ kraw,
    __bf16* __restrict__ qr, __bf16* __restrict__ kr) {
  const int row = blockIdx.x;               // 0..NROWS-1
  const int t   = threadIdx.x;              // 0..255, pairs (t, t+256)
  const int pos = row & (SEQ_LEN - 1);
  const size_t base = (size_t)row * EMBED;

  const float q_lo = qraw[base + t], q_hi = qraw[base + t + 256];
  const float k_lo = kraw[base + t], k_hi = kraw[base + t + 256];

  const float LOG_BASE = 9.2103403719761836f;     // ln(10000)
  const int i_lo = t >> 1;                        // freq index for d = t
  const int i_hi = (t + 256) >> 1;                // freq index for d = t+256
  const float ang_lo = (float)pos * expf(-LOG_BASE * (float)i_lo * (1.0f / 256.0f));
  const float ang_hi = (float)pos * expf(-LOG_BASE * (float)i_hi * (1.0f / 256.0f));
  const float c_lo = cosf(ang_lo), s_lo = sinf(ang_lo);
  const float c_hi = cosf(ang_hi), s_hi = sinf(ang_hi);

  // d < 256: rot = -x[d+256] ; d >= 256: rot = x[d-256]
  qr[base + t]       = (__bf16)(q_lo * c_lo - q_hi * s_lo);
  qr[base + t + 256] = (__bf16)(q_hi * c_hi + q_lo * s_hi);
  kr[base + t]       = (__bf16)(k_lo * c_lo - k_hi * s_lo);
  kr[base + t + 256] = (__bf16)(k_hi * c_hi + k_lo * s_hi);
}

// ---------------- Phase 2: causal flash attention --------------------------
// grid (SEQ_LEN/BM, BATCH); block 256 (8 waves).
// Wave w: S-tile rows 16*(w&3), cols 32*(w>>2); O slice cols [64*w, 64*w+64).
__global__ __launch_bounds__(256) void rope_attn_flash(
    const __bf16* __restrict__ qr, const __bf16* __restrict__ kr,
    const __bf16* __restrict__ vT, float* __restrict__ out) {
  extern __shared__ char smem[];
  __bf16* sQ  = (__bf16*)smem;                 // [BM][LDQ]
  __bf16* sK  = sQ + BM * LDQ;                 // [BN][LDQ]
  __bf16* sVt = sK + BN * LDQ;                 // [EMBED][LDV] (V transposed)
  __bf16* sP  = sVt + EMBED * LDV;             // [BM][LDP]
  float* m_s     = (float*)(sP + BM * LDP);    // [BM] running max
  float* l_s     = m_s + BM;                   // [BM] running sum
  float* alpha_s = l_s + BM;                   // [BM] exp(m_old - m_new)
  float* red_max = alpha_s + BM;               // [2][BM]
  float* red_sum = red_max + 2 * BM;           // [2][BM]

  const int qt = blockIdx.x, b = blockIdx.y;
  const int q0 = qt * BM;
  const size_t bbase  = (size_t)b * SEQ_LEN * EMBED;
  const size_t btbase = (size_t)b * EMBED * SEQ_LEN;  // vT base for batch b
  const int tid = threadIdx.x, lane = tid & 31, w = tid >> 5;
  const int rs = w & 3, cs = w >> 2;          // S-tile assignment
  const int nn = lane & 15, half = lane >> 4;

  // Stage Q tile once: async global->LDS, 16B per lane per op.
  for (int i = tid; i < BM * (EMBED / 8); i += 256) {
    int r = i >> 6, c8 = i & 63;
    async_copy_b128(&sQ[(size_t)r * LDQ + c8 * 8],
                    &qr[bbase + (size_t)(q0 + r) * EMBED + c8 * 8]);
  }
  if (tid < BM) { m_s[tid] = -3.0e38f; l_s[tid] = 0.0f; }

  v8f of[4][4];
#pragma unroll
  for (int r = 0; r < 4; ++r)
#pragma unroll
    for (int c = 0; c < 4; ++c) of[r][c] = zero_v8f();

  for (int kt = 0; kt <= qt; ++kt) {
    const int k0 = kt * BN;
    __syncthreads();   // previous iteration done with sK/sVt/sP
    // K tile [BN][EMBED] and V^T tile [EMBED][BN]: async global->LDS.
    for (int i = tid; i < BN * (EMBED / 8); i += 256) {
      int r = i >> 6, c8 = i & 63;
      async_copy_b128(&sK[(size_t)r * LDQ + c8 * 8],
                      &kr[bbase + (size_t)(k0 + r) * EMBED + c8 * 8]);
    }
    for (int i = tid; i < EMBED * (BN / 8); i += 256) {
      int d = i >> 3, c8 = i & 7;
      async_copy_b128(&sVt[(size_t)d * LDV + c8 * 8],
                      &vT[btbase + (size_t)d * SEQ_LEN + k0 + c8 * 8]);
    }
    if (kt < qt) {  // prefetch next K tile into L2 (global_prefetch_b8)
      const __bf16* nk = &kr[bbase + (size_t)(k0 + BN) * EMBED];
      __builtin_prefetch(nk + (size_t)tid * 256, 0, 1);
    }
    async_wait0();       // this wave's async copies (incl. Q on kt==0) done
    __syncthreads();     // all waves' copies visible

    // S = Q K^T over full d = 512 (16 WMMA k-steps), wave tile [16][32].
    // Software-pipelined: load kk+32 fragments before kk's WMMAs issue.
    v8f sf[2];
    sf[0] = zero_v8f(); sf[1] = zero_v8f();
    {
      v16bf a  = lds_load_a (sQ, LDQ, 16 * rs, 0, lane);
      v16bf b0 = lds_load_bt(sK, LDQ, 32 * cs,      0, lane);
      v16bf b1 = lds_load_bt(sK, LDQ, 32 * cs + 16, 0, lane);
#pragma unroll
      for (int kk = 0; kk < EMBED; kk += 32) {
        v16bf a_n = a, b0_n = b0, b1_n = b1;
        if (kk + 32 < EMBED) {
          a_n  = lds_load_a (sQ, LDQ, 16 * rs, kk + 32, lane);
          b0_n = lds_load_bt(sK, LDQ, 32 * cs,      kk + 32, lane);
          b1_n = lds_load_bt(sK, LDQ, 32 * cs + 16, kk + 32, lane);
        }
        sf[0] = wmma_bf16(a, b0, sf[0]);
        sf[1] = wmma_bf16(a, b1, sf[1]);
        a = a_n; b0 = b0_n; b1 = b1_n;
      }
    }

    // Scale + causal mask (fp32).
    float sv[2][8];
#pragma unroll
    for (int t2 = 0; t2 < 2; ++t2)
#pragma unroll
      for (int v = 0; v < 8; ++v) {
        const int qi = q0 + 16 * rs + v + 8 * half;
        const int kj = k0 + 32 * cs + 16 * t2 + nn;
        const float s = sf[t2][v] * SCALE;
        sv[t2][v] = (kj <= qi) ? s : -3.0e38f;
      }

    // Row max: reduce across the 16 lanes holding one row (masks 1,2,4,8).
    float rmax[8];
#pragma unroll
    for (int v = 0; v < 8; ++v) rmax[v] = fmaxf(sv[0][v], sv[1][v]);
#pragma unroll
    for (int m = 1; m <= 8; m <<= 1)
#pragma unroll
      for (int v = 0; v < 8; ++v)
        rmax[v] = fmaxf(rmax[v], __shfl_xor(rmax[v], m, 32));
    if (nn == 0)
#pragma unroll
      for (int v = 0; v < 8; ++v)
        red_max[cs * BM + 16 * rs + v + 8 * half] = rmax[v];
    __syncthreads();

    if (tid < BM) {
      const float mc = fmaxf(red_max[tid], red_max[BM + tid]);
      const float mo = m_s[tid];
      const float mn = fmaxf(mo, mc);
      alpha_s[tid] = __expf(mo - mn);
      m_s[tid] = mn;
    }
    __syncthreads();

    // P = exp(S - m); stage bf16 P; row-wise partial sums.
    float rsum[8] = {0.f, 0.f, 0.f, 0.f, 0.f, 0.f, 0.f, 0.f};
#pragma unroll
    for (int t2 = 0; t2 < 2; ++t2)
#pragma unroll
      for (int v = 0; v < 8; ++v) {
        const int rl = 16 * rs + v + 8 * half;
        const float p = __expf(sv[t2][v] - m_s[rl]);
        rsum[v] += p;
        sP[(size_t)rl * LDP + 32 * cs + 16 * t2 + nn] = (__bf16)p;
      }
#pragma unroll
    for (int m = 1; m <= 8; m <<= 1)
#pragma unroll
      for (int v = 0; v < 8; ++v) rsum[v] += __shfl_xor(rsum[v], m, 32);
    if (nn == 0)
#pragma unroll
      for (int v = 0; v < 8; ++v)
        red_sum[cs * BM + 16 * rs + v + 8 * half] = rsum[v];
    __syncthreads();

    if (tid < BM)
      l_s[tid] = l_s[tid] * alpha_s[tid] + red_sum[tid] + red_sum[BM + tid];

    // Rescale O accumulators (all 64 rows of this wave's d-slice).
#pragma unroll
    for (int r = 0; r < 4; ++r) {
      float al[8];
#pragma unroll
      for (int v = 0; v < 8; ++v) al[v] = alpha_s[16 * r + v + 8 * half];
#pragma unroll
      for (int c = 0; c < 4; ++c)
#pragma unroll
        for (int v = 0; v < 8; ++v) of[r][c][v] *= al[v];
    }
    __syncthreads();

    // O += P @ V  (K-dim 64 = 2 WMMA steps; wave owns cols [64w, 64w+64)).
#pragma unroll
    for (int kk2 = 0; kk2 < 2; ++kk2) {
      v16bf bfr[4];
#pragma unroll
      for (int c = 0; c < 4; ++c)
        bfr[c] = lds_load_bt(sVt, LDV, 64 * w + 16 * c, 32 * kk2, lane);
#pragma unroll
      for (int r = 0; r < 4; ++r) {
        v16bf a = lds_load_a(sP, LDP, 16 * r, 32 * kk2, lane);
#pragma unroll
        for (int c = 0; c < 4; ++c)
          of[r][c] = wmma_bf16(a, bfr[c], of[r][c]);
      }
    }
  }

  __syncthreads();
  // Epilogue: O / l -> out (fp32).
#pragma unroll
  for (int r = 0; r < 4; ++r)
#pragma unroll
    for (int c = 0; c < 4; ++c)
#pragma unroll
      for (int v = 0; v < 8; ++v) {
        const int rl = 16 * r + v + 8 * half;
        const int col = 64 * w + 16 * c + nn;
        out[bbase + (size_t)(q0 + rl) * EMBED + col] = of[r][c][v] / l_s[rl];
      }
}

// ---------------------------------------------------------------------------
extern "C" void kernel_launch(void* const* d_in, const int* in_sizes, int n_in,
                              void* d_out, int out_size, void* d_ws, size_t ws_size,
                              hipStream_t stream) {
  const float* x  = (const float*)d_in[0];
  const float* Wq = (const float*)d_in[1];
  const float* bq = (const float*)d_in[2];
  const float* Wk = (const float*)d_in[3];
  const float* bk = (const float*)d_in[4];
  const float* Wv = (const float*)d_in[5];
  const float* bv = (const float*)d_in[6];
  float* out = (float*)d_out;

  // Workspace: q_raw f32 | k_raw f32 | qr bf16 | kr bf16 | vT bf16 (117 MB).
  constexpr size_t NB = (size_t)NROWS * EMBED;   // 8,388,608 elements
  float*  qraw = (float*)d_ws;
  float*  kraw = qraw + NB;
  __bf16* qr   = (__bf16*)(kraw + NB);
  __bf16* kr   = qr + NB;
  __bf16* vT   = kr + NB;

  dim3 g1(NROWS / 64, EMBED / 64, 3);
  rope_attn_qkv_gemm<<<g1, 128, 0, stream>>>(x, Wq, bq, Wk, bk, Wv, bv,
                                             qraw, kraw, vT);
  rope_attn_rope<<<NROWS, 256, 0, stream>>>(qraw, kraw, qr, kr);

  constexpr size_t SMEM_BYTES =
      (size_t)(2 * BM * LDQ + EMBED * LDV + BM * LDP) * 2 +
      (size_t)(3 * BM + 4 * BM) * 4;
  rope_attn_flash<<<dim3(SEQ_LEN / BM, BATCH), 256, SMEM_BYTES, stream>>>(
      qr, kr, vT, out);
}
